// SeqVLADModule_33913061769334
// MI455X (gfx1250) — matrix operand order repaired and checked by reading, not compiled, not run on hardware
//
#include <hip/hip_runtime.h>
#include <hip/hip_bf16.h>
#include <cstdint>
#include <cstddef>

#define TT   16
#define BB   8
#define BTOT 128     // BB*TT
#define CIN  768
#define DDIM 256
#define KCEN 64
#define WW   28
#define HWP  784
#define PST  800     // padded row stride (784 + 16) -> K loops are multiples of 32
#define EPSN 1e-12f

typedef __attribute__((ext_vector_type(16))) _Float16 v16h;
typedef __attribute__((ext_vector_type(8)))  _Float16 v8h;
typedef __attribute__((ext_vector_type(8)))  float    v8f;

static __device__ __forceinline__ v8f wmma16(v16h a, v16h b, v8f c) {
  // v_wmma_f32_16x16x32_f16  (neg_a, A, neg_b, B, c_mod, C, reuse_a, reuse_b)
  return __builtin_amdgcn_wmma_f32_16x16x32_f16(false, a, false, b, (short)0, c,
                                                false, false);
}

// A fragment (16x32 f16, M x K): lane r=lane&15 -> row M=r; lane half hi:
//   slots j<8 : K = k0 + hi*16 + j        (contiguous 8 halves)
//   slots j>=8: K = k0 + hi*16 + j        (contiguous 8 halves)
static __device__ __forceinline__ v16h load_a_f16(const _Float16* __restrict__ row,
                                                  int k0, int hi) {
  v8h lo = *(const v8h*)(row + k0 + hi * 16);
  v8h up = *(const v8h*)(row + k0 + hi * 16 + 8);
  return __builtin_shufflevector(lo, up, 0,1,2,3,4,5,6,7,8,9,10,11,12,13,14,15);
}

// ---------------------------------------------------------------------------
// Weight pre-conversion: f32 -> f16; GRU 3x3 filters relaid out tap-major
// uzf[tap][m][ki] = U_z[m][ki][tap] so WMMA A-rows are contiguous in ki.
// ---------------------------------------------------------------------------
__global__ void k_prep(const float* __restrict__ redu_w, const float* __restrict__ share_w,
                       const float* __restrict__ uz, const float* __restrict__ ur,
                       const float* __restrict__ uh,
                       _Float16* __restrict__ whf, _Float16* __restrict__ swf,
                       _Float16* __restrict__ uzf, _Float16* __restrict__ urf,
                       _Float16* __restrict__ uhf) {
  int idx = blockIdx.x * blockDim.x + threadIdx.x;
  if (idx < DDIM * CIN)  whf[idx] = (_Float16)redu_w[idx];
  if (idx < KCEN * DDIM) swf[idx] = (_Float16)share_w[idx];
  if (idx < 9 * KCEN * KCEN) {
    int tap = idx >> 12;          // / 4096
    int m   = (idx >> 6) & 63;
    int ki  = idx & 63;
    int src = (m * KCEN + ki) * 9 + tap;
    uzf[idx] = (_Float16)uz[src];
    urf[idx] = (_Float16)ur[src];
    uhf[idx] = (_Float16)uh[src];
  }
}

__global__ void k_zero(float* __restrict__ p, int n) {
  int i = blockIdx.x * blockDim.x + threadIdx.x;
  if (i < n) p[i] = 0.f;
}

// zero the 16-float pad column of every xr / hs row so padded-K GEMMs read 0
__global__ void k_padzero(float* __restrict__ xr, float* __restrict__ hs) {
  int idx = blockIdx.x * blockDim.x + threadIdx.x;
  const int nrow_xr = BTOT * DDIM;
  const int nrow_hs = BTOT * KCEN;
  if (idx >= (nrow_xr + nrow_hs) * 16) return;
  int row = idx >> 4, c = idx & 15;
  if (row < nrow_xr) xr[(size_t)row * PST + HWP + c] = 0.f;
  else hs[(size_t)(row - nrow_xr) * PST + HWP + c] = 0.f;
}

// ---------------------------------------------------------------------------
// 1x1 conv as GEMM:  out[img][m][p] = sum_k W[m][k] * in[img][k][p] + bias[m]
// One wave per 16(M) x 16(P) tile, K in steps of 32 via v_wmma_f32_16x16x32_f16.
// ---------------------------------------------------------------------------
template <int MTOT, int KTOT, int INS, int OUTS>
__global__ void k_gemm(const float* __restrict__ in, const _Float16* __restrict__ w,
                       const float* __restrict__ bias, float* __restrict__ out) {
  const int lane = threadIdx.x & 31;
  const int gw   = (blockIdx.x * blockDim.x + threadIdx.x) >> 5;
  const int NT = HWP / 16, MT = MTOT / 16;
  if (gw >= BTOT * MT * NT) return;
  const int img = gw / (MT * NT);
  const int mt  = (gw / NT) % MT;
  const int nt  = gw % NT;
  const int r = lane & 15, hi = lane >> 4;

  const _Float16* Arow = w + (size_t)(mt * 16 + r) * KTOT;
  const float*    Xb   = in + (size_t)img * KTOT * INS + nt * 16 + r;
  v8f c = {};
  for (int k0 = 0; k0 < KTOT; k0 += 32) {
    v16h a = load_a_f16(Arow, k0, hi);
    if (k0 + 32 < KTOT)   // gfx1250 global_prefetch_b8 of next K-chunk
      __builtin_prefetch(Xb + (size_t)(k0 + 32 + hi * 16) * INS, 0, 0);
    v16h b{};
#pragma unroll
    for (int j = 0; j < 16; ++j) {   // B slot j: K = k0 + hi*16 + j, N = r
      b[j] = (_Float16)Xb[(size_t)(k0 + hi * 16 + j) * INS];
    }
    c = wmma16(a, b, c);
  }
  float* Ob = out + (size_t)img * MTOT * OUTS + nt * 16 + r;
#pragma unroll
  for (int i = 0; i < 8; ++i) {      // C VGPR i: M = mt*16 + i + hi*8, N = r
    int m = mt * 16 + i + hi * 8;
    Ob[(size_t)m * OUTS] = c[i] + bias[m];
  }
}

// ---------------------------------------------------------------------------
// ConvGRU step, kernel 1: z = sigmoid(w + U_z*h), r = sigmoid(w + U_r*h),
// also writes rh = r .* h.  3x3 SAME conv = 9 accumulated K=64 WMMA GEMMs.
// Halo handling: clamp address + 0/1 mask multiply (no predicated loads).
// ---------------------------------------------------------------------------
__global__ void k_gru_zr(const float* __restrict__ hs, const float* __restrict__ wxb,
                         const _Float16* __restrict__ uz, const _Float16* __restrict__ ur,
                         float* __restrict__ zbuf, float* __restrict__ rhbuf, int t) {
  const int lane = threadIdx.x & 31;
  const int gw   = (blockIdx.x * blockDim.x + threadIdx.x) >> 5;
  const int NT = HWP / 16, MT = KCEN / 16;
  if (gw >= BB * MT * NT) return;
  const int b  = gw / (MT * NT);
  const int mt = (gw / NT) % MT;
  const int nt = gw % NT;
  const int r = lane & 15, hi = lane >> 4;
  const int n = nt * 16 + r;
  const int py = n / WW, px = n % WW;

  const int tp = (t > 0) ? (t - 1) : 0;
  const float* hb = hs + (size_t)(b * TT + tp) * KCEN * PST;
  v8f cz = {}, cr = {};
  if (t > 0) {
#pragma unroll
    for (int tap = 0; tap < 9; ++tap) {
      const int dy = tap / 3 - 1, dx = tap % 3 - 1;
      const bool valid = ((unsigned)(py + dy) < (unsigned)WW) &&
                         ((unsigned)(px + dx) < (unsigned)WW);
      const int   n2s = valid ? (n + dy * WW + dx) : 0;  // always-legal address
      const float msk = valid ? 1.f : 0.f;
      const _Float16* azr = uz + tap * KCEN * KCEN + (mt * 16 + r) * KCEN;
      const _Float16* arr = ur + tap * KCEN * KCEN + (mt * 16 + r) * KCEN;
      for (int k0 = 0; k0 < KCEN; k0 += 32) {
        v16h bf{};
#pragma unroll
        for (int j = 0; j < 16; ++j) {
          int kk = k0 + hi * 16 + j;
          bf[j] = (_Float16)(hb[(size_t)kk * PST + n2s] * msk);
        }
        v16h a1 = load_a_f16(azr, k0, hi);
        v16h a2 = load_a_f16(arr, k0, hi);
        cz = wmma16(a1, bf, cz);
        cr = wmma16(a2, bf, cr);
      }
    }
  }
  const float* wb = wxb + (size_t)(b * TT + t) * KCEN * PST;
  float* zb = zbuf + (size_t)b * KCEN * PST;
  float* rb = rhbuf + (size_t)b * KCEN * PST;
#pragma unroll
  for (int i = 0; i < 8; ++i) {
    int m = mt * 16 + i + hi * 8;
    size_t idx = (size_t)m * PST + n;
    float wv = wb[idx];
    float z  = 1.f / (1.f + __expf(-(wv + cz[i])));
    float rr = 1.f / (1.f + __expf(-(wv + cr[i])));
    float hp = (t > 0) ? hb[idx] : 0.f;
    zb[idx] = z;
    rb[idx] = rr * hp;
  }
}

// ConvGRU step, kernel 2: hh = tanh(w + U_h*(r.*h)); h_t = (1-z)*hh + z*h_{t-1}
__global__ void k_gru_h(float* __restrict__ hs, const float* __restrict__ wxb,
                        const _Float16* __restrict__ uh,
                        const float* __restrict__ zbuf, const float* __restrict__ rhbuf,
                        int t) {
  const int lane = threadIdx.x & 31;
  const int gw   = (blockIdx.x * blockDim.x + threadIdx.x) >> 5;
  const int NT = HWP / 16, MT = KCEN / 16;
  if (gw >= BB * MT * NT) return;
  const int b  = gw / (MT * NT);
  const int mt = (gw / NT) % MT;
  const int nt = gw % NT;
  const int r = lane & 15, hi = lane >> 4;
  const int n = nt * 16 + r;
  const int py = n / WW, px = n % WW;

  const float* rb = rhbuf + (size_t)b * KCEN * PST;
  v8f ch = {};
#pragma unroll
  for (int tap = 0; tap < 9; ++tap) {
    const int dy = tap / 3 - 1, dx = tap % 3 - 1;
    const bool valid = ((unsigned)(py + dy) < (unsigned)WW) &&
                       ((unsigned)(px + dx) < (unsigned)WW);
    const int   n2s = valid ? (n + dy * WW + dx) : 0;
    const float msk = valid ? 1.f : 0.f;
    const _Float16* ah = uh + tap * KCEN * KCEN + (mt * 16 + r) * KCEN;
    for (int k0 = 0; k0 < KCEN; k0 += 32) {
      v16h bf{};
#pragma unroll
      for (int j = 0; j < 16; ++j) {
        int kk = k0 + hi * 16 + j;
        bf[j] = (_Float16)(rb[(size_t)kk * PST + n2s] * msk);
      }
      v16h a1 = load_a_f16(ah, k0, hi);
      ch = wmma16(a1, bf, ch);
    }
  }
  const int tp = (t > 0) ? (t - 1) : 0;
  const float* hb = hs + (size_t)(b * TT + tp) * KCEN * PST;
  const float* wb = wxb + (size_t)(b * TT + t) * KCEN * PST;
  const float* zb = zbuf + (size_t)b * KCEN * PST;
  float* ho = hs + (size_t)(b * TT + t) * KCEN * PST;
#pragma unroll
  for (int i = 0; i < 8; ++i) {
    int m = mt * 16 + i + hi * 8;
    size_t idx = (size_t)m * PST + n;
    float hhv = tanhf(wb[idx] + ch[i]);
    float z   = zb[idx];
    float hp  = (t > 0) ? hb[idx] : 0.f;
    ho[idx] = (1.f - z) * hhv + z * hp;
  }
}

// Softmax over K=64 centers at each (img, pixel), in place (pads untouched).
__global__ void k_softmax(float* __restrict__ hs) {
  int tid = blockIdx.x * blockDim.x + threadIdx.x;
  if (tid >= BTOT * HWP) return;
  int img = tid / HWP, p = tid % HWP;
  float* base = hs + (size_t)img * KCEN * PST + p;
  float v[KCEN];
  float mx = -3.4e38f;
#pragma unroll
  for (int k = 0; k < KCEN; ++k) { v[k] = base[(size_t)k * PST]; mx = fmaxf(mx, v[k]); }
  float s = 0.f;
#pragma unroll
  for (int k = 0; k < KCEN; ++k) { v[k] = __expf(v[k] - mx); s += v[k]; }
  float inv = 1.f / s;
#pragma unroll
  for (int k = 0; k < KCEN; ++k) base[(size_t)k * PST] = v[k] * inv;
}

// a_sum: one wave per (img, k); fold the sum over t directly via atomics.
__global__ void k_asum(const float* __restrict__ assign, float* __restrict__ asum_b) {
  int lane = threadIdx.x & 31;
  int gw   = (blockIdx.x * blockDim.x + threadIdx.x) >> 5;
  if (gw >= BTOT * KCEN) return;
  int img = gw / KCEN, k = gw % KCEN;
  const float* row = assign + ((size_t)img * KCEN + k) * PST;
  float s = 0.f;
  for (int p = lane; p < HWP; p += 32) s += row[p];
#pragma unroll
  for (int off = 16; off > 0; off >>= 1) s += __shfl_down(s, off, 32);
  if (lane == 0) atomicAdd(asum_b + (img / TT) * KCEN + k, s);
}

// VLAD: vlad[b][k][d] = sum_t sum_p assign[bt][k][p]*xr[bt][d][p]
//                       - (sum_t a_sum[bt][k]) * centers[k][d]
// K = padded 800 (pad cols are zero) -> 25 clean 32-wide WMMA chunks, no masks.
__global__ void k_vlad(const float* __restrict__ assign, const float* __restrict__ xr,
                       const float* __restrict__ asum_b, const float* __restrict__ centers,
                       float* __restrict__ vlad) {
  const int lane = threadIdx.x & 31;
  const int gw   = (blockIdx.x * blockDim.x + threadIdx.x) >> 5;
  const int MT = KCEN / 16, NTD = DDIM / 16;
  if (gw >= BB * MT * NTD) return;
  const int b  = gw / (MT * NTD);
  const int mt = (gw / NTD) % MT;
  const int nt = gw % NTD;
  const int r = lane & 15, hi = lane >> 4;

  v8f c = {};
  for (int t = 0; t < TT; ++t) {
    const float* Ab = assign + ((size_t)(b * TT + t) * KCEN + mt * 16 + r) * PST;
    const float* Bb = xr     + ((size_t)(b * TT + t) * DDIM + nt * 16 + r) * PST;
    for (int p0 = 0; p0 < PST; p0 += 32) {
      v16h a{}, bf{};
#pragma unroll
      for (int j = 0; j < 16; ++j) {
        int ka = p0 + hi * 16 + j;
        a[j] = (_Float16)Ab[ka];
        int kb = p0 + hi * 16 + j;
        bf[j] = (_Float16)Bb[kb];
      }
      c = wmma16(a, bf, c);
    }
  }
#pragma unroll
  for (int i = 0; i < 8; ++i) {
    int k = mt * 16 + i + hi * 8;
    int d = nt * 16 + r;
    vlad[((size_t)b * KCEN + k) * DDIM + d] =
        c[i] - asum_b[b * KCEN + k] * centers[(size_t)k * DDIM + d];
  }
}

// intra-norm over d for each (b,k), in place (one wave per row)
__global__ void k_norm1(float* __restrict__ vlad) {
  int lane = threadIdx.x & 31;
  int gw   = (blockIdx.x * blockDim.x + threadIdx.x) >> 5;
  if (gw >= BB * KCEN) return;
  float* row = vlad + (size_t)gw * DDIM;
  float ss = 0.f;
  for (int d = lane; d < DDIM; d += 32) { float v = row[d]; ss += v * v; }
#pragma unroll
  for (int off = 16; off > 0; off >>= 1) ss += __shfl_down(ss, off, 32);
  ss = __shfl(ss, 0, 32);
  float scale = 1.f / fmaxf(sqrtf(ss), EPSN);
  for (int d = lane; d < DDIM; d += 32) row[d] *= scale;
}

// global L2 norm per batch, writes final output [BB][KCEN*DDIM]
__global__ void k_norm2(const float* __restrict__ vlad, float* __restrict__ out) {
  __shared__ float sred[9];
  int b = blockIdx.x;
  const float* src = vlad + (size_t)b * KCEN * DDIM;
  float ss = 0.f;
  for (int i = threadIdx.x; i < KCEN * DDIM; i += 256) { float v = src[i]; ss += v * v; }
#pragma unroll
  for (int off = 16; off > 0; off >>= 1) ss += __shfl_down(ss, off, 32);
  if ((threadIdx.x & 31) == 0) sred[threadIdx.x >> 5] = ss;
  __syncthreads();
  if (threadIdx.x == 0) {
    float t = 0.f;
    for (int i = 0; i < 8; ++i) t += sred[i];
    sred[8] = 1.f / fmaxf(sqrtf(t), EPSN);
  }
  __syncthreads();
  float scale = sred[8];
  for (int i = threadIdx.x; i < KCEN * DDIM; i += 256)
    out[(size_t)b * KCEN * DDIM + i] = src[i] * scale;
}

// ---------------------------------------------------------------------------
extern "C" void kernel_launch(void* const* d_in, const int* in_sizes, int n_in,
                              void* d_out, int out_size, void* d_ws, size_t ws_size,
                              hipStream_t stream) {
  const float* x       = (const float*)d_in[0];
  const float* redu_w  = (const float*)d_in[1];
  const float* redu_b  = (const float*)d_in[2];
  const float* share_w = (const float*)d_in[3];
  const float* share_b = (const float*)d_in[4];
  const float* U_z     = (const float*)d_in[5];
  const float* U_r     = (const float*)d_in[6];
  const float* U_h     = (const float*)d_in[7];
  const float* centers = (const float*)d_in[8];
  float* out = (float*)d_out;

  char* ws = (char*)d_ws;
  size_t off = 0;
  auto alloc = [&](size_t bytes) -> char* {
    char* p = ws + off;
    off += (bytes + 255) & ~(size_t)255;
    return p;
  };
  _Float16* whf   = (_Float16*)alloc((size_t)DDIM * CIN * 2);
  _Float16* swf   = (_Float16*)alloc((size_t)KCEN * DDIM * 2);
  _Float16* uzf   = (_Float16*)alloc((size_t)9 * KCEN * KCEN * 2);
  _Float16* urf   = (_Float16*)alloc((size_t)9 * KCEN * KCEN * 2);
  _Float16* uhf   = (_Float16*)alloc((size_t)9 * KCEN * KCEN * 2);
  float* xr       = (float*)alloc((size_t)BTOT * DDIM * PST * 4);
  float* wxb      = (float*)alloc((size_t)BTOT * KCEN * PST * 4);
  float* hs       = (float*)alloc((size_t)BTOT * KCEN * PST * 4);
  float* zbuf     = (float*)alloc((size_t)BB * KCEN * PST * 4);
  float* rhbuf    = (float*)alloc((size_t)BB * KCEN * PST * 4);
  float* asum_b   = (float*)alloc((size_t)BB * KCEN * 4);
  float* vlad     = (float*)alloc((size_t)BB * KCEN * DDIM * 4);

  const int TPB = 256;
  auto blocks_for_threads = [](long long t) { return (int)((t + 255) / 256); };
  auto blocks_for_waves   = [](long long w) { return (int)((w * 32 + 255) / 256); };

  // weight conversion, asum zero, pad-column zero
  k_prep<<<blocks_for_threads((long long)DDIM * CIN), TPB, 0, stream>>>(
      redu_w, share_w, U_z, U_r, U_h, whf, swf, uzf, urf, uhf);
  k_zero<<<blocks_for_threads(BB * KCEN), TPB, 0, stream>>>(asum_b, BB * KCEN);
  k_padzero<<<blocks_for_threads((long long)(BTOT * DDIM + BTOT * KCEN) * 16),
              TPB, 0, stream>>>(xr, hs);

  // 1x1 convs as WMMA GEMMs (x has stride 784; intermediates stride 800)
  k_gemm<DDIM, CIN, HWP, PST>
      <<<blocks_for_waves((long long)BTOT * (DDIM / 16) * (HWP / 16)), TPB, 0, stream>>>(
          x, whf, redu_b, xr);
  k_gemm<KCEN, DDIM, PST, PST>
      <<<blocks_for_waves((long long)BTOT * (KCEN / 16) * (HWP / 16)), TPB, 0, stream>>>(
          xr, swf, share_b, wxb);

  // sequential ConvGRU
  const int gru_blocks = blocks_for_waves((long long)BB * (KCEN / 16) * (HWP / 16));
  for (int t = 0; t < TT; ++t) {
    k_gru_zr<<<gru_blocks, TPB, 0, stream>>>(hs, wxb, uzf, urf, zbuf, rhbuf, t);
    k_gru_h<<<gru_blocks, TPB, 0, stream>>>(hs, wxb, uhf, zbuf, rhbuf, t);
  }

  // softmax over centers, per-pixel
  k_softmax<<<blocks_for_threads((long long)BTOT * HWP), TPB, 0, stream>>>(hs);

  // a_sum folded over t
  k_asum<<<blocks_for_waves((long long)BTOT * KCEN), TPB, 0, stream>>>(hs, asum_b);

  // VLAD aggregation (t-accumulated WMMA GEMM) + center subtraction
  k_vlad<<<blocks_for_waves((long long)BB * (KCEN / 16) * (DDIM / 16)), TPB, 0, stream>>>(
      hs, xr, asum_b, centers, vlad);

  // normalizations
  k_norm1<<<blocks_for_waves((long long)BB * KCEN), TPB, 0, stream>>>(vlad);
  k_norm2<<<BB, TPB, 0, stream>>>(vlad, out);

  (void)in_sizes; (void)n_in; (void)out_size; (void)ws_size;
}
